// HadamardPackedLinear_54073638256923
// MI455X (gfx1250) — matrix operand order repaired and checked by reading, not compiled, not run on hardware
//
#include <hip/hip_runtime.h>
#include <stdint.h>

// ---------------------------------------------------------------------------
// HadamardPackedLinear for MI455X (gfx1250, wave32, WMMA)
//   y = (H x_grouped) @ W_ternary (per group) -> H-mix -> * beta
// Hadamard = Sylvester/sqrt(32): done as 5-stage lane butterfly (shfl_xor).
// Ternary matmul: v_wmma_f32_16x16x32_bf16, weights pre-unpacked to d_ws in
// exact B-fragment order.
// ---------------------------------------------------------------------------

typedef __attribute__((ext_vector_type(16))) __bf16 v16bf;
typedef __attribute__((ext_vector_type(8)))  __bf16 v8bf;
typedef __attribute__((ext_vector_type(8)))  float  v8f;

namespace hpl {
constexpr int A_DIM = 32;     // algebra dim (groups)
constexpr int IN_O  = 128;    // features per group (K total per group)
constexpr int OUT_O = 128;    // outputs per group (N total per group)
constexpr int D     = A_DIM * IN_O;   // 4096
constexpr int TOK   = 16;     // tokens per workgroup (one WMMA M tile)

// LDS layout (bf16 elements), padded against bank conflicts:
//  sx[g][t][i]: t-stride 136 (b128 A-frag loads ok, 16B aligned),
//               g-stride 2184 (= 16*136 + 8; keeps 16B alignment, and the
//               per-(t,i) lane-g b16 stores land on distinct-ish banks)
constexpr int SX_TS = 136;
constexpr int SX_GS = 2184;
//  sy[t][g][o]: g-stride 130 -> dword stride 65 == 1 mod 64 -> stage-3 reads
//               across g (lane) are conflict-free
constexpr int SY_GS = 130;
constexpr int SY_TS = A_DIM * SY_GS;  // 4160
constexpr size_t SX_BYTES  = (size_t)A_DIM * SX_GS * 2;  // 139776
constexpr size_t SY_BYTES  = (size_t)TOK * SY_TS * 2;    // 133120
constexpr size_t LDS_BYTES = SX_BYTES + SY_BYTES;        // 272896 <= 320KB

constexpr float INV_SQRT32 = 0.17677669529663687f;
}  // namespace hpl

// ---------------------------------------------------------------------------
// Pre-pass: unpack 2-bit ternary weights into bf16 B-fragments in d_ws.
// Fragment (g, ks, nt) = B tile (K = ks*32..+31 of group g, N = nt*16..+15).
// Lane layout (contiguous K, same as A): N = lane%16; lanes 0-15 hold K-half 0..15, lanes 16-31
// hold K-half 16..31; element e (VGPR v=e/2, pos=e%2) -> K = half*16 + e.
// Each lane's 16 bf16 values stored contiguously (32 B) for b128 loads.
// ---------------------------------------------------------------------------
__global__ __launch_bounds__(256) void hpl_unpack(
    const uint8_t* __restrict__ wp, __bf16* __restrict__ wf) {
  using namespace hpl;
  const int tid  = blockIdx.x * 256 + threadIdx.x;   // 0..32767
  const int lane = tid & 31;
  const int frag = tid >> 5;                         // 0..1023 = (g*4+ks)*8+nt
  const int g    = frag >> 5;
  const int ks   = (frag >> 3) & 3;
  const int nt   = frag & 7;
  const int o    = (nt << 4) + (lane & 15);
  const int half = lane >> 4;
  // packed row (g,o) is 32 bytes; K-window = ks*32 + half*16 + e, 2 bits each
  // -> 16 fields = one dword at byte offset ks*8 + half*4.
  const uint32_t W =
      *(const uint32_t*)(wp + (((g << 7) + o) << 5) + (ks << 3) + (half << 2));
  __bf16* dst = wf + ((size_t)frag << 9) + (lane << 4);
#pragma unroll
  for (int e = 0; e < 16; ++e) {
    const int shift = ((e >> 2) << 3) + ((3 - (e & 3)) << 1);
    const int code  = (int)((W >> shift) & 3u);      // {0,1,2} -> {-1,0,1}
    dst[e] = (__bf16)((float)(code - 1));
  }
}

// ---------------------------------------------------------------------------
// Main kernel: 16 tokens per WG, 256 threads (8 waves).
// ---------------------------------------------------------------------------
__device__ __forceinline__ float fht32(float v, int lane) {
#pragma unroll
  for (int m = 1; m < 32; m <<= 1) {
    const float p = __shfl_xor(v, m, 32);
    v = (lane & m) ? (p - v) : (v + p);
  }
  return v;
}

__global__ __launch_bounds__(256) void hpl_main(
    const float* __restrict__ x, const float* __restrict__ beta,
    const __bf16* __restrict__ wf, float* __restrict__ out) {
  using namespace hpl;
  extern __shared__ char smem[];
  __bf16* sx = (__bf16*)smem;
  __bf16* sy = (__bf16*)(smem + SX_BYTES);

  const int tid  = threadIdx.x;
  const int lane = tid & 31;
  const int w    = tid >> 5;
  const long tb  = (long)blockIdx.x * TOK;

  // ---- stage 1: x -> FHT over groups (lane = g) -> bf16 x_mixed in LDS ----
  {
    const float* xg = x + tb * D + lane * IN_O;
    for (int it = 0; it < (TOK * IN_O) / 8; ++it) {
      const int p = it * 8 + w;          // (t,i) pair
      const int t = p >> 7, i = p & 127;
      float v = fht32(xg[(long)t * D + i], lane) * INV_SQRT32;
      sx[lane * SX_GS + t * SX_TS + i] = (__bf16)v;   // lane is now h-index
    }
  }
  __syncthreads();

  // ---- stage 2: per-group GEMM: y_parts[g] = x_mixed[g] @ W[g]^T ----------
  {
    const int tA = lane & 15;            // A-matrix M (token)
    const int kb = (lane >> 4) << 4;     // A-matrix K base: {0..15} vs
                                         //                  {16..31}
    for (int gg = 0; gg < 4; ++gg) {
      const int g = (w << 2) + gg;       // wave owns 4 groups
      v16bf a[4];
#pragma unroll
      for (int ks = 0; ks < 4; ++ks) {   // K = ks*32 .. +31
        const __bf16* base = sx + g * SX_GS + tA * SX_TS + (ks << 5) + kb;
        const v8bf lo = *(const v8bf*)(base);        // K = kb..kb+7
        const v8bf hi = *(const v8bf*)(base + 8);    // K = kb+8..kb+15
        a[ks] = __builtin_shufflevector(lo, hi, 0, 1, 2, 3, 4, 5, 6, 7, 8, 9,
                                        10, 11, 12, 13, 14, 15);
      }
#pragma unroll
      for (int nt = 0; nt < 8; ++nt) {
        v8f c = {};
#pragma unroll
        for (int ks = 0; ks < 4; ++ks) {
          const v16bf b = *(const v16bf*)(
              wf + ((size_t)((((g << 2) | ks) << 3) | nt) << 9) + (lane << 4));
          c = __builtin_amdgcn_wmma_f32_16x16x32_bf16(
              false, a[ks], false, b, (short)0, c, false, false);
        }
        // C layout: lane -> N = lane%16, VGPR r -> M = (lane/16)*8 + r
        const int o  = (nt << 4) + (lane & 15);
        const int t0 = (lane >> 4) << 3;
#pragma unroll
        for (int r = 0; r < 8; ++r)
          sy[(t0 + r) * SY_TS + g * SY_GS + o] = (__bf16)c[r];
      }
    }
  }
  __syncthreads();

  // ---- stage 3: FHT over groups (lane = g -> h) + beta scale + store ------
  {
    for (int it = 0; it < (TOK * OUT_O) / 8; ++it) {
      const int p = it * 8 + w;
      const int t = p >> 7, o = p & 127;
      float v = fht32((float)sy[t * SY_TS + lane * SY_GS + o], lane);
      v *= INV_SQRT32 * beta[o];
      out[(tb + t) * (long)D + lane * OUT_O + o] = v;  // channel = h*128 + o
    }
  }
}

// ---------------------------------------------------------------------------
// Host entry
// inputs: d_in[0]=x f32 [B,T,4096], d_in[1]=weight_packed u8 [32,128,32],
//         d_in[2]=beta f32 [128], d_in[3]=H f32 (unused; FHT hardcoded)
// output: f32 [B,T,4096]
// workspace: 1 MB bf16 unpacked weight fragments
// ---------------------------------------------------------------------------
extern "C" void kernel_launch(void* const* d_in, const int* in_sizes, int n_in,
                              void* d_out, int out_size, void* d_ws,
                              size_t ws_size, hipStream_t stream) {
  using namespace hpl;
  const float*   x    = (const float*)d_in[0];
  const uint8_t* wp   = (const uint8_t*)d_in[1];
  const float*   beta = (const float*)d_in[2];
  float*         out  = (float*)d_out;
  __bf16*        wf   = (__bf16*)d_ws;   // 32*128*128 bf16 = 1 MB

  const int BT = in_sizes[0] / D;        // 8192 tokens

  // allow > default dynamic LDS (273 KB of the 320 KB WGP pool)
  (void)hipFuncSetAttribute((const void*)hpl_main,
                            hipFuncAttributeMaxDynamicSharedMemorySize,
                            (int)LDS_BYTES);

  // 1) unpack ternary weights into WMMA B-fragment layout (deterministic,
  //    re-run every call; ordered before main kernel on the same stream)
  hpl_unpack<<<(A_DIM * 4 * 8 * 32) / 256, 256, 0, stream>>>(wp, wf);

  // 2) main fused kernel: 16 tokens per workgroup
  hpl_main<<<BT / TOK, 256, LDS_BYTES, stream>>>(x, beta, wf, out);
}
